// GCNEncoderPyg_33062658245462
// MI455X (gfx1250) — matrix operand order repaired and checked by reading, not compiled, not run on hardware
//
#include <hip/hip_runtime.h>
#include <hip/hip_bf16.h>
#include <math.h>

#define NN   16384   // nodes
#define KIN  512     // input dim
#define KHID 256     // hidden dim
#define KOUT 128     // output dim

typedef __attribute__((ext_vector_type(16))) _Float16 v16h;
typedef __attribute__((ext_vector_type(8)))  _Float16 v8h;
typedef __attribute__((ext_vector_type(8)))  float    v8f;
typedef __attribute__((ext_vector_type(4)))  float    v4f;

// ---------------------------------------------------------------------------
// K0: degrees -> din[r] = deg>0 ? 1/sqrt(deg) : 0   (A symmetric: row sums)
// ---------------------------------------------------------------------------
__global__ __launch_bounds__(256)
void deg_kernel(const float* __restrict__ A, float* __restrict__ din) {
    __shared__ float red[256];
    const int row = blockIdx.x;
    const size_t base = (size_t)row * NN;
    float s = 0.f;
    for (int j = threadIdx.x; j < NN / 4; j += 256) {
        v4f v = *(const v4f*)(A + base + (size_t)j * 4);
        s += v.x + v.y + v.z + v.w;
    }
    red[threadIdx.x] = s;
    __syncthreads();
    for (int off = 128; off > 0; off >>= 1) {
        if (threadIdx.x < off) red[threadIdx.x] += red[threadIdx.x + off];
        __syncthreads();
    }
    if (threadIdx.x == 0) {
        float d = red[0];
        din[row] = (d > 0.f) ? rsqrtf(d) : 0.f;
    }
}

// ---------------------------------------------------------------------------
// Tiny transpose: WT[c][k] = f16(W[k][c])   (coalesced reads, c fast)
// ---------------------------------------------------------------------------
template <int K, int C>
__global__ __launch_bounds__(256)
void transpose_w_f16(const float* __restrict__ W, _Float16* __restrict__ WT) {
    int idx = blockIdx.x * 256 + threadIdx.x;     // over K*C
    int k = idx / C;
    int c = idx % C;
    WT[(size_t)c * K + k] = (_Float16)W[idx];
}

// ---------------------------------------------------------------------------
// K1/K3 (WMMA): hT[c][i] = f16( (X[i,:] . W[:,c]) * din[i] )
//   block = 256 threads (8 waves), 16 nodes per block; X row-tile staged
//   f32->f16 in LDS as A operand; WT gives contiguous B fragments; result
//   transposed through LDS so each column is one 32B store.
// ---------------------------------------------------------------------------
template <int K, int C>
__global__ __launch_bounds__(256)
void wmma_gemm_hT(const float* __restrict__ X,      // [NN][K] f32
                  const _Float16* __restrict__ WT,  // [C][K]  f16
                  const float* __restrict__ din,    // [NN]
                  _Float16* __restrict__ hT)        // [C][NN] f16
{
    constexpr int TPW = C / 128;                    // column tiles per wave
    __shared__ __align__(16) _Float16 xs[16 * K];
    __shared__ __align__(16) _Float16 zh[16 * C];
    __shared__ float dinS[16];

    const int tid  = threadIdx.x;
    const int wave = tid >> 5;
    const int lane = tid & 31;
    const int i0   = blockIdx.x * 16;

    if (tid < 16) dinS[tid] = din[i0 + tid];

    // stage X[i0..i0+15][0..K) as f16
    for (int idx = tid; idx < 16 * K / 4; idx += 256) {
        int r  = idx / (K / 4);
        int c4 = (idx % (K / 4)) * 4;
        v4f v = *(const v4f*)(X + (size_t)(i0 + r) * K + c4);
        xs[r * K + c4 + 0] = (_Float16)v.x;
        xs[r * K + c4 + 1] = (_Float16)v.y;
        xs[r * K + c4 + 2] = (_Float16)v.z;
        xs[r * K + c4 + 3] = (_Float16)v.w;
    }
    __syncthreads();

    v8f acc[TPW];
#pragma unroll
    for (int t = 0; t < TPW; ++t)
#pragma unroll
        for (int r = 0; r < 8; ++r) acc[t][r] = 0.f;

    const int mA   = lane & 15;
    const int kbA  = (lane >> 4) * 16;
    const int nSub = lane & 15;
    const int kbB  = (lane >> 4) * 16;

#pragma unroll
    for (int kk = 0; kk < K; kk += 32) {
        v8h lo = *(const v8h*)&xs[mA * K + kk + kbA];
        v8h hi = *(const v8h*)&xs[mA * K + kk + 8 + kbA];
        v16h af;
#pragma unroll
        for (int h = 0; h < 8; ++h) { af[h] = lo[h]; af[8 + h] = hi[h]; }
#pragma unroll
        for (int t = 0; t < TPW; ++t) {
            int c = (wave * TPW + t) * 16 + nSub;
            v16h bf = *(const v16h*)&WT[(size_t)c * K + kk + kbB];
            acc[t] = __builtin_amdgcn_wmma_f32_16x16x32_f16(
                false, af, false, bf, (short)0, acc[t], false, false);
        }
    }

    // scale by din[node] and transpose through LDS
    const int mBase = (lane >> 4) * 8;
#pragma unroll
    for (int t = 0; t < TPW; ++t) {
        int n = (wave * TPW + t) * 16 + nSub;
#pragma unroll
        for (int r = 0; r < 8; ++r) {
            int m = mBase + r;
            zh[m * C + n] = (_Float16)(acc[t][r] * dinS[m]);
        }
    }
    __syncthreads();
    for (int c = tid; c < C; c += 256) {
        v16h tmp;
#pragma unroll
        for (int j = 0; j < 16; ++j) tmp[j] = zh[j * C + c];
        *(v16h*)&hT[(size_t)c * NN + i0] = tmp;
    }
}

// ---------------------------------------------------------------------------
// K2/K4: out = l2norm( elu?( (A @ h) * din + bias ) ), fused per 16-row tile.
//   Double-buffered A staging (f32->f16 through LDS) overlaps the HBM
//   A-stream with WMMA; B fragments software-pipelined one K-step ahead.
// ---------------------------------------------------------------------------
template <int NCOLS, int ELU>
__global__ __launch_bounds__(256)
void agg_norm_kernel(const float* __restrict__ A,       // [NN][NN]
                     const _Float16* __restrict__ hT,   // [NCOLS][NN]
                     const float* __restrict__ din,     // [NN]
                     const float* __restrict__ bias,    // [NCOLS]
                     float* __restrict__ out)           // [NN][NCOLS]
{
    constexpr int TPW = NCOLS / 128;                    // tiles per wave (2 or 1)
    __shared__ __align__(16) _Float16 a_lds[2][16 * 128];
    __shared__ float z_lds[16 * NCOLS];
    __shared__ float norm2[16];
    __shared__ float dinS[16];
    __shared__ float bS[NCOLS];

    const int tid  = threadIdx.x;
    const int wave = tid >> 5;
    const int lane = tid & 31;
    const int m0   = blockIdx.x * 16;

    if (tid < 16) { dinS[tid] = din[m0 + tid]; norm2[tid] = 0.f; }
    if (tid < NCOLS) bS[tid] = bias[tid];

    v8f acc[TPW];
#pragma unroll
    for (int t = 0; t < TPW; ++t)
#pragma unroll
        for (int r = 0; r < 8; ++r) acc[t][r] = 0.f;

    // A staging map: thread loads rows rA and rA+8, 4 floats at ccA
    const int rA  = tid >> 5;           // 0..7
    const int ccA = (tid & 31) * 4;     // 0..124
    const float* aRow0 = A + (size_t)(m0 + rA) * NN + ccA;
    const float* aRow1 = A + (size_t)(m0 + rA + 8) * NN + ccA;

    // fragment index math (contiguous k = 16*(lane>>4) + s)
    const int mA   = lane & 15;
    const int kbA  = (lane >> 4) * 16;
    const int nSub = lane & 15;
    const int kbB  = (lane >> 4) * 16;

    const _Float16* bBase[TPW];
#pragma unroll
    for (int t = 0; t < TPW; ++t)
        bBase[t] = hT + (size_t)((wave * TPW + t) * 16 + nSub) * NN + kbB;

    // prologue: stage chunk 0 into buffer 0
    {
        v4f v0 = *(const v4f*)(aRow0);
        v4f v1 = *(const v4f*)(aRow1);
        _Float16* d0 = &a_lds[0][rA * 128 + ccA];
        _Float16* d1 = &a_lds[0][(rA + 8) * 128 + ccA];
        d0[0] = (_Float16)v0.x; d0[1] = (_Float16)v0.y;
        d0[2] = (_Float16)v0.z; d0[3] = (_Float16)v0.w;
        d1[0] = (_Float16)v1.x; d1[1] = (_Float16)v1.y;
        d1[2] = (_Float16)v1.z; d1[3] = (_Float16)v1.w;
    }
    // preload B fragments for the first K-step
    v16h bf[TPW];
#pragma unroll
    for (int t = 0; t < TPW; ++t) bf[t] = *(const v16h*)(bBase[t]);

    for (int kc = 0; kc < NN; kc += 128) {
        const int cur = (kc >> 7) & 1;
        __syncthreads();                       // a_lds[cur] ready for all waves
        const bool more = (kc + 128) < NN;
        v4f n0, n1;
        if (more) {                            // issue next A chunk early
            n0 = *(const v4f*)(aRow0 + kc + 128);
            n1 = *(const v4f*)(aRow1 + kc + 128);
        }
#pragma unroll
        for (int kk = 0; kk < 128; kk += 32) {
            v8h lo = *(const v8h*)&a_lds[cur][mA * 128 + kk + kbA];
            v8h hi = *(const v8h*)&a_lds[cur][mA * 128 + kk + 8 + kbA];
            v16h af;
#pragma unroll
            for (int h = 0; h < 8; ++h) { af[h] = lo[h]; af[8 + h] = hi[h]; }

            int knext = kc + kk + 32;          // pipeline B one step ahead
            knext = (knext < NN) ? knext : 0;
            v16h bf_n[TPW];
#pragma unroll
            for (int t = 0; t < TPW; ++t)
                bf_n[t] = *(const v16h*)(bBase[t] + knext);
#pragma unroll
            for (int t = 0; t < TPW; ++t)
                acc[t] = __builtin_amdgcn_wmma_f32_16x16x32_f16(
                    false, af, false, bf[t], (short)0, acc[t], false, false);
#pragma unroll
            for (int t = 0; t < TPW; ++t) bf[t] = bf_n[t];
        }
        if (more) {                            // convert+store next chunk
            _Float16* d0 = &a_lds[cur ^ 1][rA * 128 + ccA];
            _Float16* d1 = &a_lds[cur ^ 1][(rA + 8) * 128 + ccA];
            d0[0] = (_Float16)n0.x; d0[1] = (_Float16)n0.y;
            d0[2] = (_Float16)n0.z; d0[3] = (_Float16)n0.w;
            d1[0] = (_Float16)n1.x; d1[1] = (_Float16)n1.y;
            d1[2] = (_Float16)n1.z; d1[3] = (_Float16)n1.w;
        }
    }

    // epilogue: scale + bias (+ELU) -> LDS (C layout: M=8*(L>=16)+r, N=L&15)
    const int mBase = (lane >> 4) * 8;
#pragma unroll
    for (int t = 0; t < TPW; ++t) {
        int n = (wave * TPW + t) * 16 + nSub;
#pragma unroll
        for (int r = 0; r < 8; ++r) {
            int m = mBase + r;
            float v = acc[t][r] * dinS[m] + bS[n];
            if (ELU) v = (v > 0.f) ? v : expm1f(v);
            z_lds[m * NCOLS + n] = v;
        }
    }
    __syncthreads();

    // fused row L2 norm: 16 threads per row
    constexpr int CPT = NCOLS / 16;
    const int row = tid >> 4;
    const int c0  = tid & 15;
    float part = 0.f;
#pragma unroll
    for (int j = 0; j < CPT; ++j) {
        float z = z_lds[row * NCOLS + c0 + j * 16];
        part += z * z;
    }
    atomicAdd(&norm2[row], part);              // ds_add_f32
    __syncthreads();

    const float rinv = 1.f / fmaxf(sqrtf(norm2[row]), 1e-12f);
#pragma unroll
    for (int j = 0; j < CPT; ++j) {
        int col = c0 + j * 16;
        out[(size_t)(m0 + row) * NCOLS + col] = z_lds[row * NCOLS + col] * rinv;
    }
}

// ---------------------------------------------------------------------------
extern "C" void kernel_launch(void* const* d_in, const int* in_sizes, int n_in,
                              void* d_out, int out_size, void* d_ws, size_t ws_size,
                              hipStream_t stream) {
    const float* x  = (const float*)d_in[0];   // [NN][KIN]
    const float* A  = (const float*)d_in[1];   // [NN][NN]
    const float* W1 = (const float*)d_in[2];   // [KIN][KHID]
    const float* b1 = (const float*)d_in[3];   // [KHID]
    const float* W2 = (const float*)d_in[4];   // [KHID][KOUT]
    const float* b2 = (const float*)d_in[5];   // [KOUT]

    float* z2_out = (float*)d_out;                         // [NN][KOUT]
    float* z1_out = z2_out + (size_t)NN * KOUT;            // [NN][KHID]

    char* ws = (char*)d_ws;
    size_t off = 0;
    float*    din = (float*)(ws + off);      off += (size_t)NN * 4;
    _Float16* h1T = (_Float16*)(ws + off);   off += (size_t)KHID * NN * 2;
    _Float16* h2T = (_Float16*)(ws + off);   off += (size_t)KOUT * NN * 2;
    _Float16* W1T = (_Float16*)(ws + off);   off += (size_t)KIN * KHID * 2;
    _Float16* W2T = (_Float16*)(ws + off);   off += (size_t)KHID * KOUT * 2;

    // degrees / normalization factors
    deg_kernel<<<NN, 256, 0, stream>>>(A, din);
    // weight transposes (f16)
    transpose_w_f16<KIN, KHID><<<(KIN * KHID) / 256, 256, 0, stream>>>(W1, W1T);
    transpose_w_f16<KHID, KOUT><<<(KHID * KOUT) / 256, 256, 0, stream>>>(W2, W2T);
    // layer 1: h1T = ((x@W1)*din)^T in f16   (WMMA)
    wmma_gemm_hT<KIN, KHID><<<NN / 16, 256, 0, stream>>>(x, W1T, din, h1T);
    // z1 = l2norm(elu(A@h1 * din + b1))      (WMMA, pipelined)
    agg_norm_kernel<KHID, 1><<<NN / 16, 256, 0, stream>>>(A, h1T, din, b1, z1_out);
    // layer 2: h2T = ((z1@W2)*din)^T in f16  (WMMA)
    wmma_gemm_hT<KHID, KOUT><<<NN / 16, 256, 0, stream>>>(z1_out, W2T, din, h2T);
    // z2 = l2norm(A@h2 * din + b2)           (WMMA, pipelined)
    agg_norm_kernel<KOUT, 0><<<NN / 16, 256, 0, stream>>>(A, h2T, din, b2, z2_out);
}
